// RWL_69355131896530
// MI455X (gfx1250) — matrix operand order repaired and checked by reading, not compiled, not run on hardware
//
#include <hip/hip_runtime.h>
#include <hip/hip_bf16.h>

// ---------------------------------------------------------------------------
// Types for CDNA5 WMMA (gfx1250, wave32)
// ---------------------------------------------------------------------------
typedef __attribute__((ext_vector_type(16))) __bf16   v16bf;
typedef __attribute__((ext_vector_type(8)))  float    v8f;
typedef __attribute__((ext_vector_type(4)))  unsigned u32x4;

union FragBF {
    v16bf  v;
    u32x4  q[2];
    unsigned u[8];
};

__device__ __forceinline__ unsigned short f2bf(float f) {
    unsigned u = __float_as_uint(f);
    u += 0x7fffu + ((u >> 16) & 1u);     // round-to-nearest-even
    return (unsigned short)(u >> 16);
}
__device__ __forceinline__ float bf2f(unsigned short h) {
    return __uint_as_float(((unsigned)h) << 16);
}
__device__ __forceinline__ v8f wmma_bf16(const FragBF& a, const FragBF& b, v8f c) {
    return __builtin_amdgcn_wmma_f32_16x16x32_bf16(
        false, a.v, false, b.v, (short)0, c, false, false);
}

// ---------------------------------------------------------------------------
// Prep kernels: fp32 -> bf16 convert / transpose (weights into N x K layout)
// ---------------------------------------------------------------------------
__global__ void cvt_bf16_kernel(const float* __restrict__ src,
                                unsigned short* __restrict__ dst, int n) {
    int i = blockIdx.x * blockDim.x + threadIdx.x;
    if (i < n) dst[i] = f2bf(src[i]);
}

// src is K x N (row-major), dst is N x Kpad (row-major), zero-padded k >= K
__global__ void transpose_bf16_kernel(const float* __restrict__ src,
                                      unsigned short* __restrict__ dst,
                                      int K, int N, int Kpad) {
    int i = blockIdx.x * blockDim.x + threadIdx.x;
    int total = N * Kpad;
    if (i >= total) return;
    int n = i / Kpad, k = i - n * Kpad;
    dst[i] = (k < K) ? f2bf(src[(size_t)k * N + n]) : (unsigned short)0;
}

// Build state0 = [s, x[:, :127]] (1024 x 128) and st = [s, x, pad] (1024 x 160)
__global__ void build_states_kernel(const float* __restrict__ tptr,
                                    const float* __restrict__ x,
                                    unsigned short* __restrict__ state0,
                                    unsigned short* __restrict__ stb) {
    int i = blockIdx.x * blockDim.x + threadIdx.x;
    if (i >= 1024 * 160) return;
    int b = i / 160, c = i - b * 160;
    float tv = *tptr;
    float v = (c == 0) ? tv : ((c <= 128) ? x[(size_t)b * 128 + (c - 1)] : 0.f);
    stb[i] = f2bf(v);
    if (c < 128) {
        float v0 = (c == 0) ? tv : x[(size_t)b * 128 + (c - 1)];
        state0[(size_t)b * 128 + c] = f2bf(v0);
    }
}

// ---------------------------------------------------------------------------
// Generic bf16 WMMA GEMM: C[M x N] = act(A[M x K] @ Wt^T + bias)
//   A  : bf16, M x K row-major
//   Wt : bf16, N x K row-major (i.e. W transposed)
//   doTanh: outH = tanh(v) (bf16), outD = 1 - tanh^2 (bf16, optional)
//   else  : outF = v (fp32)
// block = 128 threads (4 waves), each wave = 32x32 tile, block tile = 64x64
// ---------------------------------------------------------------------------
__global__ __launch_bounds__(128)
void gemm_bf16_wmma(const unsigned short* __restrict__ A,
                    const unsigned short* __restrict__ Wt,
                    const float* __restrict__ bias,
                    unsigned short* __restrict__ outH,
                    unsigned short* __restrict__ outD,
                    float* __restrict__ outF,
                    int M, int N, int K, int doTanh) {
    const int lane = threadIdx.x & 31;
    const int wv   = threadIdx.x >> 5;     // 0..3
    const int hl   = lane >> 4;            // half-wave
    const int lm   = lane & 15;
    const int m0 = blockIdx.y * 64 + (wv >> 1) * 32;
    const int n0 = blockIdx.x * 64 + (wv & 1) * 32;

    v8f acc[2][2] = {};
    for (int kb = 0; kb < K; kb += 32) {
        FragBF a[2], b[2];
#pragma unroll
        for (int ms = 0; ms < 2; ++ms) {
            const unsigned short* base = A + (size_t)(m0 + ms * 16 + lm) * K + kb;
            a[ms].q[0] = *(const u32x4*)(base + hl * 8);
            a[ms].q[1] = *(const u32x4*)(base + 16 + hl * 8);
        }
#pragma unroll
        for (int ns = 0; ns < 2; ++ns) {
            const unsigned short* base = Wt + (size_t)(n0 + ns * 16 + lm) * K + kb;
            b[ns].q[0] = *(const u32x4*)(base + hl * 8);
            b[ns].q[1] = *(const u32x4*)(base + 16 + hl * 8);
        }
#pragma unroll
        for (int ms = 0; ms < 2; ++ms)
#pragma unroll
            for (int ns = 0; ns < 2; ++ns)
                acc[ms][ns] = wmma_bf16(a[ms], b[ns], acc[ms][ns]);
    }

#pragma unroll
    for (int ms = 0; ms < 2; ++ms)
#pragma unroll
        for (int ns = 0; ns < 2; ++ns) {
            int n = n0 + ns * 16 + lm;
            float bv = bias ? bias[n] : 0.f;
#pragma unroll
            for (int r = 0; r < 8; ++r) {
                int m = m0 + ms * 16 + 8 * hl + r;
                float v = acc[ms][ns][r] + bv;
                if (doTanh) {
                    float h = tanhf(v);
                    outH[(size_t)m * N + n] = f2bf(h);
                    if (outD) outD[(size_t)m * N + n] = f2bf(1.f - h * h);
                } else {
                    outF[(size_t)m * N + n] = v;
                }
            }
        }
}

// ---------------------------------------------------------------------------
// Branch head: g[b] = e1[b,:] . bdw + bdb   (448-dot per sample)
// ---------------------------------------------------------------------------
__global__ void branch_head_kernel(const unsigned short* __restrict__ e1,
                                   const float* __restrict__ bdw,
                                   const float* __restrict__ bdb,
                                   float* __restrict__ gout) {
    int b = blockIdx.x * blockDim.x + threadIdx.x;
    if (b >= 1024) return;
    float s = 0.f;
    for (int c = 0; c < 448; ++c)
        s += bf2f(e1[(size_t)b * 448 + c]) * bdw[c];
    gout[b] = s + bdb[0];
}

// ---------------------------------------------------------------------------
// Jacobian-trace kernel: one workgroup per sample b (256 thr = 8 waves).
//   Stage A: G[n][i] = sum_c gw2[n,c] * d2[c] * gdw[c,i]  -> 960x128 bf16 in LDS
//   Stage B: tr = sum_{i,n} [ (gw0[1:,:]*d0) @ gw1 ]_{i,n} * d1[n] * G[n][i]
//   dp[b] = g[b] - p[b] * tr
// Each wave owns one 16-row i-band (mt = wave id) and sweeps n-tiles two at a
// time, so each scaled A fragment feeds 2 WMMAs. Diagonal scalings (d0,d2) are
// folded into the A operand with native packed bf16 multiplies (v_pk_mul_bf16).
// Dynamic LDS: 960*128*2 + 16 bytes (CDNA5: up to 320 KB per workgroup)
// ---------------------------------------------------------------------------
__global__ __launch_bounds__(256)
void jac_trace_kernel(const unsigned short* __restrict__ gw0b,   // 128 x 1024
                      const unsigned short* __restrict__ gw1Tb,  // 960 x 1024
                      const unsigned short* __restrict__ gw2b,   // 960 x 896
                      const unsigned short* __restrict__ gdwTb,  // 128 x 896
                      const unsigned short* __restrict__ d0b,    // B x 1024
                      const unsigned short* __restrict__ d1b,    // B x 960
                      const unsigned short* __restrict__ d2b,    // B x 896
                      const float* __restrict__ gbuf,            // B
                      const float* __restrict__ p,               // B
                      float* __restrict__ dp_out) {
    extern __shared__ unsigned short smem[];
    unsigned short* G = smem;                       // 960*128 bf16
    float* red = (float*)(smem + 960 * 128);
    const int b    = blockIdx.x;
    const int lane = threadIdx.x & 31;
    const int wv   = threadIdx.x >> 5;              // 0..7 == i-tile (mt)
    const int hl   = lane >> 4;
    const int lm   = lane & 15;

    if (threadIdx.x == 0) *red = 0.f;

    const unsigned short* d0row = d0b + (size_t)b * 1024;
    const unsigned short* d1row = d1b + (size_t)b * 960;
    const unsigned short* d2row = d2b + (size_t)b * 896;

    // ---- Stage A: G^T tiles (M = i band = wv, N = n : 60 tiles), K = 896 ----
    {
        const unsigned short* arow = gdwTb + (size_t)(wv * 16 + lm) * 896;
        for (int j = 0; j < 30; ++j) {
            const int nt0 = 2 * j, nt1 = 2 * j + 1;
            v8f acc0 = {}, acc1 = {};
            for (int kb = 0; kb < 896; kb += 32) {
                FragBF a, w, s, b0, b1;
                // A = gdwT[i][c] * d2[c]  (packed bf16 multiply)
                w.q[0] = *(const u32x4*)(arow + kb + hl * 8);
                w.q[1] = *(const u32x4*)(arow + kb + 16 + hl * 8);
                s.q[0] = *(const u32x4*)(d2row + kb + hl * 8);
                s.q[1] = *(const u32x4*)(d2row + kb + 16 + hl * 8);
                a.v = w.v * s.v;                      // -> v_pk_mul_bf16 x8
                // B = gw2 (N x K row-major), two adjacent n-tiles
                const unsigned short* bbase0 = gw2b + (size_t)(nt0 * 16 + lm) * 896 + kb;
                const unsigned short* bbase1 = gw2b + (size_t)(nt1 * 16 + lm) * 896 + kb;
                b0.q[0] = *(const u32x4*)(bbase0 + hl * 8);
                b0.q[1] = *(const u32x4*)(bbase0 + 16 + hl * 8);
                b1.q[0] = *(const u32x4*)(bbase1 + hl * 8);
                b1.q[1] = *(const u32x4*)(bbase1 + 16 + hl * 8);
                acc0 = wmma_bf16(a, b0, acc0);
                acc1 = wmma_bf16(a, b1, acc1);
            }
            // Store to G[n][i]: lane's 8 values are i = wv*16 + 8*hl + r (contiguous)
            u32x4 pk0, pk1;
#pragma unroll
            for (int r = 0; r < 4; ++r) {
                pk0[r] = (unsigned)f2bf(acc0[2 * r]) | ((unsigned)f2bf(acc0[2 * r + 1]) << 16);
                pk1[r] = (unsigned)f2bf(acc1[2 * r]) | ((unsigned)f2bf(acc1[2 * r + 1]) << 16);
            }
            *(u32x4*)&G[(size_t)(nt0 * 16 + lm) * 128 + wv * 16 + 8 * hl] = pk0;
            *(u32x4*)&G[(size_t)(nt1 * 16 + lm) * 128 + wv * 16 + 8 * hl] = pk1;
        }
    }
    __syncthreads();

    // ---- Stage B: T1 tiles (M = i band = wv, N = n : 60 tiles), K = 1024 ----
    float part = 0.f;
    {
        const int irow = wv * 16 + lm;                    // row of T0
        const unsigned short* arow = gw0b + (size_t)(irow + 1) * 1024;
        const bool valid = (irow < 127);                  // T0 row 127 is zero pad
        for (int j = 0; j < 30; ++j) {
            const int nt0 = 2 * j, nt1 = 2 * j + 1;
            v8f acc0 = {}, acc1 = {};
            for (int kb = 0; kb < 1024; kb += 32) {
                FragBF a, w, s, b0, b1;
                if (valid) {
                    w.q[0] = *(const u32x4*)(arow + kb + hl * 8);
                    w.q[1] = *(const u32x4*)(arow + kb + 16 + hl * 8);
                    s.q[0] = *(const u32x4*)(d0row + kb + hl * 8);
                    s.q[1] = *(const u32x4*)(d0row + kb + 16 + hl * 8);
                    a.v = w.v * s.v;                      // -> v_pk_mul_bf16 x8
                } else {
                    a.q[0] = u32x4{0, 0, 0, 0};
                    a.q[1] = u32x4{0, 0, 0, 0};
                }
                const unsigned short* bbase0 = gw1Tb + (size_t)(nt0 * 16 + lm) * 1024 + kb;
                const unsigned short* bbase1 = gw1Tb + (size_t)(nt1 * 16 + lm) * 1024 + kb;
                b0.q[0] = *(const u32x4*)(bbase0 + hl * 8);
                b0.q[1] = *(const u32x4*)(bbase0 + 16 + hl * 8);
                b1.q[0] = *(const u32x4*)(bbase1 + hl * 8);
                b1.q[1] = *(const u32x4*)(bbase1 + 16 + hl * 8);
                acc0 = wmma_bf16(a, b0, acc0);
                acc1 = wmma_bf16(a, b1, acc1);
            }
            // tr contribution: lane has n fixed, i = wv*16 + 8*hl + r
#pragma unroll
            for (int half_tile = 0; half_tile < 2; ++half_tile) {
                const v8f& acc = half_tile ? acc1 : acc0;
                int n = (2 * j + half_tile) * 16 + lm;
                float d1v = bf2f(d1row[n]);
                u32x4 gq = *(const u32x4*)&G[(size_t)n * 128 + wv * 16 + 8 * hl];
                const unsigned short* gs = (const unsigned short*)&gq;
                float ssum = 0.f;
#pragma unroll
                for (int r = 0; r < 8; ++r) ssum += acc[r] * bf2f(gs[r]);
                part += ssum * d1v;
            }
        }
    }

    // reduce within wave, then across waves via LDS atomic
    for (int off = 16; off > 0; off >>= 1)
        part += __shfl_down(part, off, 32);
    if (lane == 0) atomicAdd(red, part);
    __syncthreads();
    if (threadIdx.x == 0)
        dp_out[b] = gbuf[b] - p[b] * (*red);
}

// ---------------------------------------------------------------------------
// Host launcher
// ---------------------------------------------------------------------------
extern "C" void kernel_launch(void* const* d_in, const int* in_sizes, int n_in,
                              void* d_out, int out_size, void* d_ws, size_t ws_size,
                              hipStream_t stream) {
    const float* pseudo_t = (const float*)d_in[0];
    const float* x   = (const float*)d_in[1];
    const float* p   = (const float*)d_in[3];
    const float* gw0 = (const float*)d_in[4];
    const float* gb0 = (const float*)d_in[5];
    const float* gw1 = (const float*)d_in[6];
    const float* gb1 = (const float*)d_in[7];
    const float* gw2 = (const float*)d_in[8];
    const float* gb2 = (const float*)d_in[9];
    const float* gdw = (const float*)d_in[10];
    const float* gdb = (const float*)d_in[11];
    const float* bw0 = (const float*)d_in[12];
    const float* bb0 = (const float*)d_in[13];
    const float* bw1 = (const float*)d_in[14];
    const float* bb1 = (const float*)d_in[15];
    const float* bdw = (const float*)d_in[16];
    const float* bdb = (const float*)d_in[17];

    float* dxdt = (float*)d_out;           // 1024 x 128
    float* gout = dxdt + 1024 * 128;       // 1024
    float* dpdt = gout + 1024;             // 1024

    char* ws = (char*)d_ws;
    size_t off = 0;
    auto alloc = [&](size_t bytes) -> void* {
        void* ptr = ws + off;
        off = (off + bytes + 255) & ~(size_t)255;
        return ptr;
    };
    unsigned short* gw0_bf = (unsigned short*)alloc((size_t)128 * 1024 * 2);
    unsigned short* gw0T   = (unsigned short*)alloc((size_t)1024 * 128 * 2);
    unsigned short* gw1T   = (unsigned short*)alloc((size_t)960 * 1024 * 2);
    unsigned short* gw2_bf = (unsigned short*)alloc((size_t)960 * 896 * 2);
    unsigned short* gw2T   = (unsigned short*)alloc((size_t)896 * 960 * 2);
    unsigned short* gdwT   = (unsigned short*)alloc((size_t)128 * 896 * 2);
    unsigned short* bw0T   = (unsigned short*)alloc((size_t)512 * 160 * 2);
    unsigned short* bw1T   = (unsigned short*)alloc((size_t)448 * 512 * 2);
    unsigned short* st0    = (unsigned short*)alloc((size_t)1024 * 128 * 2);
    unsigned short* stb    = (unsigned short*)alloc((size_t)1024 * 160 * 2);
    unsigned short* h0     = (unsigned short*)alloc((size_t)1024 * 1024 * 2);
    unsigned short* dd0    = (unsigned short*)alloc((size_t)1024 * 1024 * 2);
    unsigned short* h1     = (unsigned short*)alloc((size_t)1024 * 960 * 2);
    unsigned short* dd1    = (unsigned short*)alloc((size_t)1024 * 960 * 2);
    unsigned short* h2     = (unsigned short*)alloc((size_t)1024 * 896 * 2);
    unsigned short* dd2    = (unsigned short*)alloc((size_t)1024 * 896 * 2);
    unsigned short* e0     = (unsigned short*)alloc((size_t)1024 * 512 * 2);
    unsigned short* e1     = (unsigned short*)alloc((size_t)1024 * 448 * 2);

    auto cdiv = [](int a, int b) { return (a + b - 1) / b; };

    // ---- prep: bf16 converts & transposes ----
    cvt_bf16_kernel<<<cdiv(128 * 1024, 256), 256, 0, stream>>>(gw0, gw0_bf, 128 * 1024);
    cvt_bf16_kernel<<<cdiv(960 * 896, 256), 256, 0, stream>>>(gw2, gw2_bf, 960 * 896);
    transpose_bf16_kernel<<<cdiv(1024 * 128, 256), 256, 0, stream>>>(gw0, gw0T, 128, 1024, 128);
    transpose_bf16_kernel<<<cdiv(960 * 1024, 256), 256, 0, stream>>>(gw1, gw1T, 1024, 960, 1024);
    transpose_bf16_kernel<<<cdiv(896 * 960, 256), 256, 0, stream>>>(gw2, gw2T, 960, 896, 960);
    transpose_bf16_kernel<<<cdiv(128 * 896, 256), 256, 0, stream>>>(gdw, gdwT, 896, 128, 896);
    transpose_bf16_kernel<<<cdiv(512 * 160, 256), 256, 0, stream>>>(bw0, bw0T, 129, 512, 160);
    transpose_bf16_kernel<<<cdiv(448 * 512, 256), 256, 0, stream>>>(bw1, bw1T, 512, 448, 512);
    build_states_kernel<<<cdiv(1024 * 160, 256), 256, 0, stream>>>(pseudo_t, x, st0, stb);

    // ---- forward GRN (WMMA GEMMs) ----
    gemm_bf16_wmma<<<dim3(1024 / 64, 1024 / 64), 128, 0, stream>>>(
        st0, gw0T, gb0, h0, dd0, nullptr, 1024, 1024, 128, 1);
    gemm_bf16_wmma<<<dim3(960 / 64, 1024 / 64), 128, 0, stream>>>(
        h0, gw1T, gb1, h1, dd1, nullptr, 1024, 960, 1024, 1);
    gemm_bf16_wmma<<<dim3(896 / 64, 1024 / 64), 128, 0, stream>>>(
        h1, gw2T, gb2, h2, dd2, nullptr, 1024, 896, 960, 1);
    gemm_bf16_wmma<<<dim3(128 / 64, 1024 / 64), 128, 0, stream>>>(
        h2, gdwT, gdb, nullptr, nullptr, dxdt, 1024, 128, 896, 0);

    // ---- branch MLP ----
    gemm_bf16_wmma<<<dim3(512 / 64, 1024 / 64), 128, 0, stream>>>(
        stb, bw0T, bb0, e0, nullptr, nullptr, 1024, 512, 160, 1);
    gemm_bf16_wmma<<<dim3(448 / 64, 1024 / 64), 128, 0, stream>>>(
        e0, bw1T, bb1, e1, nullptr, nullptr, 1024, 448, 512, 1);
    branch_head_kernel<<<cdiv(1024, 128), 128, 0, stream>>>(e1, bdw, bdb, gout);

    // ---- Jacobian trace: one workgroup per sample, 240KB LDS for G ----
    const size_t smem_bytes = (size_t)960 * 128 * 2 + 16;
    jac_trace_kernel<<<1024, 256, smem_bytes, stream>>>(
        gw0_bf, gw1T, gw2_bf, gdwT, dd0, dd1, dd2, gout, p, dpdt);
}
